// SimpleRetention_2817498546685
// MI455X (gfx1250) — matrix operand order repaired and checked by reading, compiled, not run on hardware
//
#include <hip/hip_runtime.h>

// ---------------------------------------------------------------------------
// SimpleRetention on MI455X (gfx1250), wave32 + WMMA bf16 16x16x32.
// B=4, S=2048, H=256, gamma=0.9. Output complex64 (interleaved float2).
// Stage B: 32-row n-tile per block (2 WMMA row-blocks), 32-wide m-tiles,
// 8 waves splitting the H=256 reduction, fused decay in the shared reduce.
// Tile staging uses the CDNA5 async Global->LDS DMA path when available.
// ---------------------------------------------------------------------------

#define B_ 4
#define S_ 2048
#define H_ 256
#define WIN_ 1024                      // gamma^1024 == 0 in fp32 -> exact truncation
#define L2G_ (-0.15200309344504995f)   // log2(0.9)

typedef __attribute__((ext_vector_type(16))) __bf16          bf16x16;
typedef __attribute__((ext_vector_type(8)))  float           v8f;
typedef __attribute__((ext_vector_type(16))) unsigned short  us16;
typedef __attribute__((ext_vector_type(4)))  int             v4i;

// ---- CDNA5 async Global->LDS staging (guarded; falls back to sync copy) ----
#if defined(__has_builtin)
#  if __has_builtin(__builtin_amdgcn_global_load_async_to_lds_b128) && \
      __has_builtin(__builtin_amdgcn_s_wait_asynccnt)
#    define USE_ASYNC_LDS 1
#  endif
#endif
#ifndef USE_ASYNC_LDS
#  define USE_ASYNC_LDS 0
#endif

typedef __attribute__((address_space(1))) v4i* gp_v4i;   // global (AS1) int4*
typedef __attribute__((address_space(3))) v4i* lp_v4i;   // LDS    (AS3) int4*

__device__ __forceinline__ void cp_b128(const unsigned short* g, unsigned short* l) {
#if USE_ASYNC_LDS
  // ISA: LDS aperture flat-address[31:0] == wave-relative LDS byte offset.
  __builtin_amdgcn_global_load_async_to_lds_b128(
      (gp_v4i)(unsigned long long)(size_t)g,
      (lp_v4i)(unsigned int)(size_t)l, 0, 0);
#else
  *(uint4*)l = *(const uint4*)g;
#endif
}

__device__ __forceinline__ void async_fence() {
#if USE_ASYNC_LDS
  __builtin_amdgcn_s_wait_asynccnt(0);
#endif
}

// cold-path software round-to-nearest-even f32->bf16 (projection kernel)
__device__ __forceinline__ unsigned short f2bf(float f) {
  unsigned int u = __float_as_uint(f);
  u += 0x7FFFu + ((u >> 16) & 1u);
  return (unsigned short)(u >> 16);
}

// hot-path: pack two f32 -> two bf16 in one dword (v_cvt_pk_bf16_f32 if present)
__device__ __forceinline__ unsigned int pack_bf2(float a, float b) {
#if defined(__has_builtin) && __has_builtin(__builtin_amdgcn_cvt_pk_bf16_f32)
  return __builtin_bit_cast(unsigned int, __builtin_amdgcn_cvt_pk_bf16_f32(a, b));
#else
  unsigned short lo = __builtin_bit_cast(unsigned short, (__bf16)a);
  unsigned short hi = __builtin_bit_cast(unsigned short, (__bf16)b);
  return (unsigned int)lo | ((unsigned int)hi << 16);
#endif
}

__device__ __forceinline__ v8f wmma_bf16(bf16x16 a, bf16x16 b, v8f c) {
  return __builtin_amdgcn_wmma_f32_16x16x32_bf16(false, a, false, b,
                                                 (short)0, c, false, false);
}

// Two contiguous 8-element (16B) runs -> 16-bf16 WMMA operand
__device__ __forceinline__ bf16x16 load_pair16(const unsigned short* p0,
                                               const unsigned short* p1) {
  us16 u;
  ((uint4*)&u)[0] = *(const uint4*)p0;
  ((uint4*)&u)[1] = *(const uint4*)p1;
  return __builtin_bit_cast(bf16x16, u);
}

// ---------------------------------------------------------------------------
// Stage A: projections + complex rotation (fp32 math, bf16 outputs).
//   Qc = q*cos((s+1)th), Qs = q*sin, Kc = k*cos, Ks = -k*sin
// V is written TRANSPOSED: gVT[b][h][s], so stage-B B-operands are contiguous.
// ---------------------------------------------------------------------------
__global__ __launch_bounds__(H_)
void retnet_proj_rot(const float* __restrict__ X,
                     const float* __restrict__ Wq,
                     const float* __restrict__ Wk,
                     const float* __restrict__ Wv,
                     const float* __restrict__ theta,
                     unsigned short* __restrict__ gQc,
                     unsigned short* __restrict__ gQs,
                     unsigned short* __restrict__ gKc,
                     unsigned short* __restrict__ gKs,
                     unsigned short* __restrict__ gVT) {
  __shared__ float xrow[H_];
  const int row = blockIdx.x;          // b*S + s
  const int s   = row & (S_ - 1);
  const int b   = row >> 11;           // S_ == 2048
  const int h   = threadIdx.x;
  xrow[h] = X[(size_t)row * H_ + h];
  __syncthreads();
  float q = 0.f, k = 0.f, v = 0.f;
#pragma unroll 4
  for (int t = 0; t < H_; ++t) {
    const float x = xrow[t];
    q = fmaf(x, Wq[t * H_ + h], q);
    k = fmaf(x, Wk[t * H_ + h], k);
    v = fmaf(x, Wv[t * H_ + h], v);
  }
  float sn, cs;
  __sincosf((float)(s + 1) * theta[h], &sn, &cs);
  const size_t o = (size_t)row * H_ + h;
  gQc[o] = f2bf(q * cs);
  gQs[o] = f2bf(q * sn);
  gKc[o] = f2bf(k * cs);
  gKs[o] = f2bf(-k * sn);
  gVT[((size_t)b * H_ + h) * S_ + s] = f2bf(v);
}

// ---------------------------------------------------------------------------
// Stage B
// ---------------------------------------------------------------------------
__global__ __launch_bounds__(256)
void retnet_attn(const unsigned short* __restrict__ gQc,
                 const unsigned short* __restrict__ gQs,
                 const unsigned short* __restrict__ gKc,
                 const unsigned short* __restrict__ gKs,
                 const unsigned short* __restrict__ gVT,
                 float* __restrict__ out) {
  const int n0   = blockIdx.x * 32;    // 32-row n-tile (2 WMMA row-blocks)
  const int b    = blockIdx.y;
  const int tid  = threadIdx.x;
  const int lane = tid & 31;
  const int wid  = tid >> 5;
  const int h0   = wid * 32;
  const int l16  = lane & 15;
  const bool hi  = lane >= 16;
  const int kA0  = hi ? 8 : 0;   // A-operand first K run (ISA 16-bit A 16x32 layout)
  const int kB0  = hi ? 8 : 0;   // B-operand first K run (same layout as A)
  const int Mb   = hi ? 8 : 0;

  __shared__ unsigned short sKc[32 * H_];        // 16 KB
  __shared__ unsigned short sKs[32 * H_];        // 16 KB
  __shared__ unsigned short sVT[H_ * 32];        // 16 KB, [h][m] transposed
  __shared__ float sPart[8][2][16][32][2];       // per-wave att partials, 64 KB
  __shared__ unsigned int sAttP[2][2][256];      // [rowblk][plane], 4 KB

  // Q A-operands straight from global (once per block), two row-blocks
  bf16x16 Aqc[2], Aqs[2];
#pragma unroll
  for (int rb = 0; rb < 2; ++rb) {
    const size_t qrow = ((size_t)b * S_ + (size_t)(n0 + rb * 16 + l16)) * H_ + h0;
    Aqc[rb] = load_pair16(gQc + qrow + kA0, gQc + qrow + kA0 + 16);
    Aqs[rb] = load_pair16(gQs + qrow + kA0, gQs + qrow + kA0 + 16);
  }

  v8f acc[2][2][2];                    // [rowblk][h-tile][re/im]
#pragma unroll
  for (int rb = 0; rb < 2; ++rb)
#pragma unroll
    for (int ct = 0; ct < 2; ++ct)
#pragma unroll
      for (int t = 0; t < 2; ++t) acc[rb][ct][t] = (v8f){};

  int mStart = n0 + 32 - WIN_;
  if (mStart < 0) mStart = 0;
  mStart &= ~31;

  for (int m0 = mStart; m0 < n0 + 32; m0 += 32) {
    __syncthreads();  // previous iteration fully consumed
    { // ---- stage tiles via async Global->LDS (VGPR-free when available) ----
      const size_t koff = ((size_t)b * S_ + m0) * H_;
      const unsigned short* srcKc = gKc + koff;
      const unsigned short* srcKs = gKs + koff;
#pragma unroll
      for (int j = 0; j < 4; ++j) {
        const int i = (tid + j * 256) * 8;     // 8 ushorts per b128
        cp_b128(srcKc + i, sKc + i);
        cp_b128(srcKs + i, sKs + i);
      }
      // transposed V: thread tid copies row h=tid (32 bf16, contiguous)
      const unsigned short* srcV = gVT + ((size_t)b * H_ + tid) * S_ + m0;
#pragma unroll
      for (int j = 0; j < 4; ++j) cp_b128(srcV + j * 8, sVT + tid * 32 + j * 8);
      async_fence();                           // ASYNCcnt -> 0 before publish
    }
    __syncthreads();

    // ---- QK^T partials over this wave's 32-wide h-slice: 16 WMMAs ----
#pragma unroll
    for (int ct = 0; ct < 2; ++ct) {
      const unsigned short* krc = sKc + (ct * 16 + l16) * H_ + h0 + kB0;
      const unsigned short* krs = sKs + (ct * 16 + l16) * H_ + h0 + kB0;
      const bf16x16 Bkc = load_pair16(krc, krc + 16);         // shared by both rb
      const bf16x16 Bks = load_pair16(krs, krs + 16);
      const int Nc = ct * 16 + l16;
#pragma unroll
      for (int rb = 0; rb < 2; ++rb) {
        v8f z = {};
        const v8f t0 = wmma_bf16(Aqc[rb], Bkc, z);
        const v8f t1 = wmma_bf16(Aqs[rb], Bks, z);
        const v8f re = t0 - t1;                                         // QcKc - QsKs
        const v8f im = wmma_bf16(Aqc[rb], Bks, wmma_bf16(Aqs[rb], Bkc, z)); // QcKs + QsKc
#pragma unroll
        for (int r = 0; r < 8; ++r) {
          *(float2*)&sPart[wid][rb][Mb + r][Nc][0] = make_float2(re[r], im[r]);
        }
      }
    }
    __syncthreads();

    // ---- fused: cross-wave reduce + decay + bf16 pack (shared work) ----
    {
      const float4* pp = (const float4*)&sPart[0][0][0][0][0];
#pragma unroll
      for (int half = 0; half < 2; ++half) {
        const int q    = tid + half * 256;     // float4 index within one wave slice
        const int rb   = q >> 8;
        const int pidx = q & 255;              // complex-pair group
        const int Mr   = (pidx * 2) >> 5;
        const int Kr   = (pidx * 2) & 31;
        float4 s = pp[q];
#pragma unroll
        for (int w = 1; w < 8; ++w) {
          const float4 t = pp[w * 512 + q];
          s.x += t.x; s.y += t.y; s.z += t.z; s.w += t.w;
        }
        const int  d0i = (n0 + rb * 16 + Mr) - (m0 + Kr);
        const float d0 = (d0i >= 0) ? exp2f((float)d0i * L2G_) : 0.f;
        const float d1 = (d0i >= 1) ? exp2f((float)(d0i - 1) * L2G_) : 0.f;
        sAttP[rb][0][pidx] = pack_bf2(s.x * d0, s.z * d1);   // re plane
        sAttP[rb][1][pidx] = pack_bf2(s.y * d0, s.w * d1);   // im plane
      }
    }
    __syncthreads();

    // ---- att @ V: 8 WMMAs (V B-operands shared by both row-blocks) ----
    const unsigned short* v0p = sVT + (h0 + l16) * 32 + kB0;
    const unsigned short* v1p = sVT + (h0 + 16 + l16) * 32 + kB0;
    const bf16x16 Bv0 = load_pair16(v0p, v0p + 16);
    const bf16x16 Bv1 = load_pair16(v1p, v1p + 16);
#pragma unroll
    for (int rb = 0; rb < 2; ++rb) {
      const unsigned short* aRe = (const unsigned short*)&sAttP[rb][0][0] + l16 * 32;
      const unsigned short* aIm = (const unsigned short*)&sAttP[rb][1][0] + l16 * 32;
      const bf16x16 Are = load_pair16(aRe + kA0, aRe + kA0 + 16);
      const bf16x16 Aim = load_pair16(aIm + kA0, aIm + kA0 + 16);
      acc[rb][0][0] = wmma_bf16(Are, Bv0, acc[rb][0][0]);
      acc[rb][0][1] = wmma_bf16(Aim, Bv0, acc[rb][0][1]);
      acc[rb][1][0] = wmma_bf16(Are, Bv1, acc[rb][1][0]);
      acc[rb][1][1] = wmma_bf16(Aim, Bv1, acc[rb][1][1]);
    }
  }

  // ---- write complex64 output (interleaved re,im) ----
  float2* o2 = (float2*)out;
#pragma unroll
  for (int rb = 0; rb < 2; ++rb) {
#pragma unroll
    for (int r = 0; r < 8; ++r) {
      const size_t rowo =
          ((size_t)b * S_ + (size_t)(n0 + rb * 16 + Mb + r)) * H_;
      o2[rowo + h0 + l16]      = make_float2(acc[rb][0][0][r], acc[rb][0][1][r]);
      o2[rowo + h0 + 16 + l16] = make_float2(acc[rb][1][0][r], acc[rb][1][1][r]);
    }
  }
}

// ---------------------------------------------------------------------------
extern "C" void kernel_launch(void* const* d_in, const int* in_sizes, int n_in,
                              void* d_out, int out_size, void* d_ws, size_t ws_size,
                              hipStream_t stream) {
  const float* X     = (const float*)d_in[0];
  const float* Wq    = (const float*)d_in[1];
  const float* Wk    = (const float*)d_in[2];
  const float* Wv    = (const float*)d_in[3];
  const float* theta = (const float*)d_in[4];

  const size_t N = (size_t)B_ * S_ * H_;
  unsigned short* gQc = (unsigned short*)d_ws;
  unsigned short* gQs = gQc + N;
  unsigned short* gKc = gQs + N;
  unsigned short* gKs = gKc + N;
  unsigned short* gVT = gKs + N;      // total 5*N*2 = 20 MB of workspace

  retnet_proj_rot<<<B_ * S_, H_, 0, stream>>>(X, Wq, Wk, Wv, theta,
                                              gQc, gQs, gKc, gKs, gVT);

  dim3 grid(S_ / 32, B_);
  retnet_attn<<<grid, 256, 0, stream>>>(gQc, gQs, gKc, gKs, gVT, (float*)d_out);
}
